// MultiHeadAttentionQuantum_65481071396056
// MI455X (gfx1250) — matrix operand order repaired and checked by reading, not compiled, not run on hardware
//
#include <hip/hip_runtime.h>

// ---------------------------------------------------------------------------
// Quantum multi-head attention, MI455X (gfx1250, wave32, WMMA).
//   B=2, S=1024, E=1024, H=128, DK=8
//   q   = x @ Wq^T + bq                      (bf16 WMMA 16x16x32, 16x64/wave)
//   out = cos(q + phi)                       (fused epilogue)
//   attn = softmax(out @ out^T / sqrt(DK))   (f32 WMMA 16x16x4, flash-style)
//   ctx  = attn @ out                        (f32 WMMA 16x16x4)
//   y    = ctx @ Wo^T + bo                   (bf16 WMMA 16x16x32)
// k/v projections are dead in the reference math and are skipped.
// Attention stages out[b,h] (32 KB) into LDS via GLOBAL_LOAD_ASYNC_TO_LDS.
// ---------------------------------------------------------------------------

typedef __attribute__((ext_vector_type(16))) __bf16 v16bf;
typedef __attribute__((ext_vector_type(8)))  float  v8f;
typedef __attribute__((ext_vector_type(2)))  float  v2f;

#define BQ   2
#define SQ   1024
#define EQ   1024
#define HQ   128
#define DKQ  8

// ===========================================================================
// GEMM:  Y[M,N] = X[M,K] @ W[N,K]^T  (+ bias; mode 0 additionally cos(.+phi))
// bf16 WMMA 16x16x32.  Each wave owns a 16x64 strip: one A fragment is
// reused against 4 B fragments (4 accumulators) per K-step.
// ===========================================================================
__global__ void __launch_bounds__(256)
gemm_bt_kernel(const float* __restrict__ X, const float* __restrict__ W,
               const float* __restrict__ bias, const float* __restrict__ phi,
               float* __restrict__ Y, int M, int N, int K, int mode)
{
    const int lane = threadIdx.x & 31;
    const int wave = threadIdx.x >> 5;
    const int ln   = lane & 15;       // L % 16
    const int g    = lane >> 4;       // half-wave select

    const int tilesN = N >> 6;                        // 64-wide wave strips
    const int tile   = blockIdx.x * 8 + wave;
    if (tile >= (M >> 4) * tilesN) return;            // wave-uniform
    const int tm = (tile / tilesN) << 4;
    const int tn = (tile % tilesN) << 6;

    const int m = tm + ln;            // A-fragment row for this lane

    v8f acc[4] = {v8f{}, v8f{}, v8f{}, v8f{}};
    for (int kt = 0; kt < K; kt += 32) {
        // ---- A fragment: 16-bit A 16x32 layout.
        // lane L (half g): K runs [kt+8g, +8) and [kt+16+8g, +8), row m.
        const float* xr = X + (size_t)m * K + kt;
        float4 f0 = *(const float4*)(xr + g * 8);
        float4 f1 = *(const float4*)(xr + g * 8 + 4);
        float4 f2 = *(const float4*)(xr + 16 + g * 8);
        float4 f3 = *(const float4*)(xr + 16 + g * 8 + 4);
        float ta[16] = {f0.x,f0.y,f0.z,f0.w, f1.x,f1.y,f1.z,f1.w,
                        f2.x,f2.y,f2.z,f2.w, f3.x,f3.y,f3.z,f3.w};
        v16bf a;
#pragma unroll
        for (int j = 0; j < 16; ++j) a[j] = (__bf16)ta[j];

        // ---- 4 B fragments: 16-bit B 32x16 layout.
        // lane L (half g): K runs [kt+8g, +8) and [kt+16+8g, +8), col n.  B[k][n] = W[n][k].
#pragma unroll
        for (int t = 0; t < 4; ++t) {
            const int n = tn + t * 16 + ln;
            const float* wr = W + (size_t)n * K + kt;
            float4 w0 = *(const float4*)(wr + g * 8);
            float4 w1 = *(const float4*)(wr + g * 8 + 4);
            float4 w2 = *(const float4*)(wr + 16 + g * 8);
            float4 w3 = *(const float4*)(wr + 16 + g * 8 + 4);
            float tb[16] = {w0.x,w0.y,w0.z,w0.w, w1.x,w1.y,w1.z,w1.w,
                            w2.x,w2.y,w2.z,w2.w, w3.x,w3.y,w3.z,w3.w};
            v16bf bfrag;
#pragma unroll
            for (int j = 0; j < 16; ++j) bfrag[j] = (__bf16)tb[j];

            acc[t] = __builtin_amdgcn_wmma_f32_16x16x32_bf16(
                         false, a, false, bfrag, (short)0, acc[t], false, false);
        }
    }

    // ---- Epilogue.  C layout: lane holds col n, rows tm + v + 8g.
#pragma unroll
    for (int t = 0; t < 4; ++t) {
        const int n  = tn + t * 16 + ln;
        const float bn = bias[n];
        const float pn = (mode == 0) ? phi[n] : 0.0f;
#pragma unroll
        for (int v = 0; v < 8; ++v) {
            const int row = tm + v + 8 * g;
            float val = acc[t][v] + bn;
            if (mode == 0) val = __cosf(val + pn);
            Y[(size_t)row * N + n] = val;
        }
    }
}

// ===========================================================================
// Flash attention per (b,h).  out[b,h] (1024x8 f32 = 32 KB) staged in LDS
// via GLOBAL_LOAD_ASYNC_TO_LDS_B128 (ASYNCcnt-tracked, no VGPR round-trip).
// 16 waves; each wave owns 4 query blocks of 16 rows.  Per key block of 16:
//   S = Q K^T / sqrt(8)   -> 2x v_wmma_f32_16x16x4_f32
//   online softmax        -> __shfl_xor row reductions in C layout
//   acc += P V            -> 4x v_wmma_f32_16x16x4_f32 (P relaid via LDS)
// ===========================================================================
__global__ void __launch_bounds__(512)
attn_kernel(const float* __restrict__ qout, float* __restrict__ ctx)
{
    __shared__ float kv[SQ][DKQ];          // 32 KB: out[b,h] resident
    __shared__ float pbuf[16][16][16];     // 16 KB: per-wave P re-layout

    const int b = blockIdx.x / HQ;
    const int h = blockIdx.x % HQ;
    const float* base = qout + (size_t)b * SQ * EQ + h * DKQ;

    // ---- Stage out[b,h] into LDS, 2 rows of 8 floats per thread, via the
    // CDNA5 async LDS path.  INST_OFFSET is added to both the global and the
    // LDS address, so one address pair covers a 32B row with offset 0 / 16.
    for (int i = threadIdx.x; i < SQ; i += 512) {
        const float*   g32 = base + (size_t)i * EQ;
        unsigned       l32 = (unsigned)(uintptr_t)(&kv[i][0]);  // low 32b = LDS addr
        asm volatile("global_load_async_to_lds_b128 %0, %1, off offset:0"
                     :: "v"(l32), "v"(g32) : "memory");
        asm volatile("global_load_async_to_lds_b128 %0, %1, off offset:16"
                     :: "v"(l32), "v"(g32) : "memory");
    }
    asm volatile("s_wait_asynccnt 0x0" ::: "memory");
    __syncthreads();

    const int lane = threadIdx.x & 31;
    const int wave = threadIdx.x >> 5;
    const int ln   = lane & 15;
    const int g    = lane >> 4;
    const float rs = 0.3535533905932738f;   // 1/sqrt(DK)

    for (int qi = 0; qi < 4; ++qi) {
        const int qb = (wave * 4 + qi) * 16;

        // Q fragments (A 16x4 f32 layout: row ln, K = 4c + 2g + {0,1}),
        // pre-scaled by 1/sqrt(DK).
        v2f qa[2];
#pragma unroll
        for (int c = 0; c < 2; ++c) {
            const int k0 = c * 4 + 2 * g;
            qa[c].x = kv[qb + ln][k0]     * rs;
            qa[c].y = kv[qb + ln][k0 + 1] * rs;
        }

        float mrow[8], lrow[8];
        v8f acc = {};
#pragma unroll
        for (int v = 0; v < 8; ++v) { mrow[v] = -1e30f; lrow[v] = 0.0f; }

        for (int kb = 0; kb < SQ; kb += 16) {
            // ---- scores tile: S = Q K^T  (B[k][n] = kv[kb+n][k])
            v8f s = {};
#pragma unroll
            for (int c = 0; c < 2; ++c) {
                const int k0 = c * 4 + 2 * g;
                v2f bk;
                bk.x = kv[kb + ln][k0];
                bk.y = kv[kb + ln][k0 + 1];
                s = __builtin_amdgcn_wmma_f32_16x16x4_f32(
                        false, qa[c], false, bk, (short)0, s, false, false);
            }

            // ---- online softmax (row = v + 8g; cols live in lanes of half g)
            float pv[8];
#pragma unroll
            for (int v = 0; v < 8; ++v) {
                float rmax = s[v];
                rmax = fmaxf(rmax, __shfl_xor(rmax, 1, 32));
                rmax = fmaxf(rmax, __shfl_xor(rmax, 2, 32));
                rmax = fmaxf(rmax, __shfl_xor(rmax, 4, 32));
                rmax = fmaxf(rmax, __shfl_xor(rmax, 8, 32));
                const float mn   = fmaxf(mrow[v], rmax);
                const float corr = __expf(mrow[v] - mn);
                const float p    = __expf(s[v] - mn);
                float rsum = p;
                rsum += __shfl_xor(rsum, 1, 32);
                rsum += __shfl_xor(rsum, 2, 32);
                rsum += __shfl_xor(rsum, 4, 32);
                rsum += __shfl_xor(rsum, 8, 32);
                lrow[v] = lrow[v] * corr + rsum;
                mrow[v] = mn;
                acc[v] *= corr;
                pv[v]   = p;
            }

            // ---- P: C layout -> LDS -> A 16x4 layout; acc += P @ V
#pragma unroll
            for (int v = 0; v < 8; ++v) pbuf[wave][v + 8 * g][ln] = pv[v];

#pragma unroll
            for (int c = 0; c < 4; ++c) {
                const int k0 = c * 4 + 2 * g;
                v2f pa, bv;
                pa.x = pbuf[wave][ln][k0];
                pa.y = pbuf[wave][ln][k0 + 1];
                // B[k][n] = V[kb+k][n], cols >= DK padded with zero
                bv.x = (ln < DKQ) ? kv[kb + k0][ln]     : 0.0f;
                bv.y = (ln < DKQ) ? kv[kb + k0 + 1][ln] : 0.0f;
                acc = __builtin_amdgcn_wmma_f32_16x16x4_f32(
                          false, pa, false, bv, (short)0, acc, false, false);
            }
        }

        // ---- epilogue: ctx[b, row, h*DK + ln] = acc / l   (ln < DK only)
        if (ln < DKQ) {
#pragma unroll
            for (int v = 0; v < 8; ++v) {
                const int row = qb + v + 8 * g;
                ctx[((size_t)b * SQ + row) * EQ + h * DKQ + ln] =
                    acc[v] / lrow[v];
            }
        }
    }
}

// ===========================================================================
extern "C" void kernel_launch(void* const* d_in, const int* in_sizes, int n_in,
                              void* d_out, int out_size, void* d_ws, size_t ws_size,
                              hipStream_t stream)
{
    const float* x   = (const float*)d_in[0];
    const float* Wq  = (const float*)d_in[1];
    const float* bq  = (const float*)d_in[2];
    // d_in[3..6] = Wk, bk, Wv, bv : dead in the reference math
    const float* phi = (const float*)d_in[7];   // [H,DK] flat == [E]
    const float* Wo  = (const float*)d_in[8];
    const float* bo  = (const float*)d_in[9];
    float* y = (float*)d_out;

    float* outbuf = (float*)d_ws;                           // B*S*E f32 (8 MB)
    float* ctxbuf = outbuf + (size_t)BQ * SQ * EQ;          // B*S*E f32 (8 MB)

    const int M = BQ * SQ;                                  // 2048
    const int tiles = (M / 16) * (EQ / 64);                 // 2048 wave strips
    const int gemmBlocks = tiles / 8;                       // 256

    // 1) q = x @ Wq^T + bq ; out = cos(q + phi)   [fused]
    gemm_bt_kernel<<<gemmBlocks, 256, 0, stream>>>(
        x, Wq, bq, phi, outbuf, M, EQ, EQ, /*mode=*/0);

    // 2) flash attention per (b,h)
    attn_kernel<<<BQ * HQ, 512, 0, stream>>>(outbuf, ctxbuf);

    // 3) y = ctx @ Wo^T + bo
    gemm_bt_kernel<<<gemmBlocks, 256, 0, stream>>>(
        ctxbuf, Wo, bo, nullptr, y, M, EQ, EQ, /*mode=*/1);
}
